// DOMINANT_MODEL_58729382805869
// MI455X (gfx1250) — compile-verified, matrix operand order checked by reading, not run on hardware
//
#include <hip/hip_runtime.h>
#include <hip/hip_bf16.h>
#include <cstdint>

// ---------------------------------------------------------------------------
// DOMINANT graph autoencoder for MI455X (gfx1250).
// All activations/weights consumed by GEMMs are pre-converted to bf16, so the
// WMMA inner loop is pure loads + v_wmma_f32_16x16x32_bf16 (f32 accumulate),
// with zero conversion VALU in the hot path. Aggregation (gather/scatter) is
// L2-resident streaming work on the f32 GEMM outputs.
// ---------------------------------------------------------------------------

typedef __attribute__((ext_vector_type(16))) __bf16 v16bf;
typedef __attribute__((ext_vector_type(8)))  float  v8f;

#define NNODE 16384
#define FIN   512
#define HID   128

static __device__ __forceinline__ unsigned short f2bf(float f) {
  union { float f; unsigned u; } v; v.f = f;
  unsigned r = v.u + 0x7FFFu + ((v.u >> 16) & 1u);   // round-to-nearest-even
  return (unsigned short)(r >> 16);
}

union Frag { v16bf v; uint4 q[2]; };   // 16 bf16 = 32 bytes = 2x uint4

// ---------------------------------------------------------------------------
// small utility kernels
// ---------------------------------------------------------------------------
__global__ void k_fill(float* p, float val, int n) {
  int i = blockIdx.x * blockDim.x + threadIdx.x;
  if (i < n) p[i] = val;
}

__global__ void k_deg_edges(const int* __restrict__ dst, float* __restrict__ deg, int E) {
  int e = blockIdx.x * blockDim.x + threadIdx.x;
  if (e < E) atomicAdd(&deg[dst[e]], 1.0f);
}

__global__ void k_dinv(const float* __restrict__ deg, float* __restrict__ dinv, int n) {
  int i = blockIdx.x * blockDim.x + threadIdx.x;
  if (i < n) dinv[i] = rsqrtf(fmaxf(deg[i], 1.0f));
}

// plain f32 -> bf16 conversion (vectorized 4-wide)
__global__ void k_cvt_bf16(const float* __restrict__ in, unsigned short* __restrict__ out,
                           int total4) {
  int i = blockIdx.x * blockDim.x + threadIdx.x;
  if (i < total4) {
    float4 v = ((const float4*)in)[i];
    ushort4 o;
    o.x = f2bf(v.x); o.y = f2bf(v.y); o.z = f2bf(v.z); o.w = f2bf(v.w);
    ((ushort4*)out)[i] = o;
  }
}

// Wt[o*K + k] = bf16(W[k*Fout + o])   (W is [K x Fout] row-major f32)
__global__ void k_transpose_bf16(const float* __restrict__ W, unsigned short* __restrict__ Wt,
                                 int K, int Fout) {
  int idx = blockIdx.x * blockDim.x + threadIdx.x;
  if (idx < K * Fout) {
    int k = idx / Fout, o = idx - k * Fout;
    Wt[(size_t)o * K + k] = f2bf(W[idx]);
  }
}

// ---------------------------------------------------------------------------
// C[M x Nc] = A[M x K] @ B^T, A/B bf16 row-major ([M x K] / [Nc x K]), C f32.
// One wave computes a 16x64 strip (4 accumulators) so the A fragment is
// reused across 4 WMMAs. 8 waves/block -> 128x64 super-tile.
// Requires M % 128 == 0, Nc % 64 == 0, K % 32 == 0 (K % 128 for full unroll).
// ---------------------------------------------------------------------------
__global__ void __launch_bounds__(256)
k_gemm_abT_bf16(const unsigned short* __restrict__ A, const unsigned short* __restrict__ B,
                float* __restrict__ C, int M, int Nc, int K, long long ldc) {
  const int lane = threadIdx.x & 31;
  const int wave = threadIdx.x >> 5;                 // 0..7
  const int r    = lane & 15;
  const int half = lane >> 4;

  const int tileM  = blockIdx.y * 128 + wave * 16;
  const int tileN0 = blockIdx.x * 64;

  const unsigned short* arow = A + (size_t)(tileM + r) * K;
  const unsigned short* brow0 = B + (size_t)(tileN0 + 0  + r) * K;
  const unsigned short* brow1 = B + (size_t)(tileN0 + 16 + r) * K;
  const unsigned short* brow2 = B + (size_t)(tileN0 + 32 + r) * K;
  const unsigned short* brow3 = B + (size_t)(tileN0 + 48 + r) * K;

  v8f acc0 = {}, acc1 = {}, acc2 = {}, acc3 = {};

#pragma unroll 4
  for (int k0 = 0; k0 < K; k0 += 32) {
    // Fragment K-packing per ISA: lane(r,half) holds K in
    // {k0+8h..k0+8h+7} U {k0+16+8h..k0+16+8h+7}  -> two 16-byte chunks.
    const int o0 = k0 + 8 * half;
    const int o1 = k0 + 16 + 8 * half;

    Frag fa, f0, f1, f2, f3;
    fa.q[0] = *(const uint4*)(arow  + o0); fa.q[1] = *(const uint4*)(arow  + o1);
    f0.q[0] = *(const uint4*)(brow0 + o0); f0.q[1] = *(const uint4*)(brow0 + o1);
    f1.q[0] = *(const uint4*)(brow1 + o0); f1.q[1] = *(const uint4*)(brow1 + o1);
    f2.q[0] = *(const uint4*)(brow2 + o0); f2.q[1] = *(const uint4*)(brow2 + o1);
    f3.q[0] = *(const uint4*)(brow3 + o0); f3.q[1] = *(const uint4*)(brow3 + o1);

    acc0 = __builtin_amdgcn_wmma_f32_16x16x32_bf16(false, fa.v, false, f0.v,
                                                   (short)0, acc0, false, false);
    acc1 = __builtin_amdgcn_wmma_f32_16x16x32_bf16(false, fa.v, false, f1.v,
                                                   (short)0, acc1, false, false);
    acc2 = __builtin_amdgcn_wmma_f32_16x16x32_bf16(false, fa.v, false, f2.v,
                                                   (short)0, acc2, false, false);
    acc3 = __builtin_amdgcn_wmma_f32_16x16x32_bf16(false, fa.v, false, f3.v,
                                                   (short)0, acc3, false, false);
  }

  // C/D layout: lanes 0-15 -> N = lane, M = vgpr; lanes 16-31 -> N = lane-16, M = vgpr+8
  const int mbase = tileM + 8 * half;
  float* crow = C + (size_t)mbase * (size_t)ldc + tileN0 + r;
#pragma unroll
  for (int i = 0; i < 8; ++i) {
    float* p = crow + (size_t)i * (size_t)ldc;
    p[0]  = acc0[i];
    p[16] = acc1[i];
    p[32] = acc2[i];
    p[48] = acc3[i];
  }
}

// ---------------------------------------------------------------------------
// GCN aggregation: out = D^-1/2 (A + I) D^-1/2 (t), then bias + ReLU
// ---------------------------------------------------------------------------
__global__ void k_agg_self(const float* __restrict__ t, const float* __restrict__ dinv,
                           float* __restrict__ out, int total, int logF) {
  int idx = blockIdx.x * blockDim.x + threadIdx.x;
  if (idx < total) {
    float dv = dinv[idx >> logF];
    out[idx] = dv * dv * t[idx];
  }
}

__global__ void __launch_bounds__(256)
k_agg_edges(const float* __restrict__ t, const int* __restrict__ src,
            const int* __restrict__ dst, const float* __restrict__ dinv,
            float* __restrict__ out, int E, int F) {
  int e = blockIdx.x * 8 + (threadIdx.x >> 5);
  if (e >= E) return;
  int lane = threadIdx.x & 31;
  int s = src[e], d = dst[e];
  float w = dinv[s] * dinv[d];
  const float4* ts = (const float4*)(t + (size_t)s * F);
  float* od = out + (size_t)d * F;
  for (int f = lane; f * 4 < F; f += 32) {
    float4 v = ts[f];
    atomicAdd(od + f * 4 + 0, w * v.x);
    atomicAdd(od + f * 4 + 1, w * v.y);
    atomicAdd(od + f * 4 + 2, w * v.z);
    atomicAdd(od + f * 4 + 3, w * v.w);
  }
}

// bias + ReLU, result written as bf16 (next consumer is always a bf16 GEMM)
__global__ void k_bias_relu_bf16(const float* __restrict__ in, const float* __restrict__ b,
                                 unsigned short* __restrict__ out, int total, int Fmask) {
  int idx = blockIdx.x * blockDim.x + threadIdx.x;
  if (idx < total) out[idx] = f2bf(fmaxf(in[idx] + b[idx & Fmask], 0.0f));
}

// bias + ReLU in place (f32, for the final attr_recon output)
__global__ void k_bias_relu_f32(float* __restrict__ out, const float* __restrict__ b,
                                int total, int Fmask) {
  int idx = blockIdx.x * blockDim.x + threadIdx.x;
  if (idx < total) out[idx] = fmaxf(out[idx] + b[idx & Fmask], 0.0f);
}

// ---------------------------------------------------------------------------
extern "C" void kernel_launch(void* const* d_in, const int* in_sizes, int n_in,
                              void* d_out, int out_size, void* d_ws, size_t ws_size,
                              hipStream_t stream) {
  (void)n_in; (void)out_size; (void)ws_size;

  const float* x      = (const float*)d_in[0];
  const int*   edges  = (const int*)d_in[1];
  const int    E      = in_sizes[1] / 2;
  const int*   src    = edges;          // edge_index[0]
  const int*   dst    = edges + E;      // edge_index[1]
  const float* W_enc0 = (const float*)d_in[2];
  const float* b_enc0 = (const float*)d_in[3];
  const float* W_enc1 = (const float*)d_in[4];
  const float* b_enc1 = (const float*)d_in[5];
  const float* W_sd0  = (const float*)d_in[6];
  const float* b_sd0  = (const float*)d_in[7];
  const float* W_sd1  = (const float*)d_in[8];
  const float* b_sd1  = (const float*)d_in[9];
  const float* W_ad0  = (const float*)d_in[10];
  const float* b_ad0  = (const float*)d_in[11];
  const float* W_ad1  = (const float*)d_in[12];
  const float* b_ad1  = (const float*)d_in[13];

  // workspace layout (f32 region, then bf16 region)
  float* wsf = (float*)d_ws;
  size_t off = 0;
  float* tmp  = wsf + off; off += (size_t)NNODE * FIN;   // GEMM result (max N x 512)
  float* agg  = wsf + off; off += (size_t)NNODE * HID;   // aggregation scratch (f32)
  float* deg  = wsf + off; off += NNODE;
  float* dinv = wsf + off; off += NNODE;
  unsigned short* wsh = (unsigned short*)(wsf + off);
  size_t hoff = 0;
  unsigned short* xb    = wsh + hoff; hoff += (size_t)NNODE * FIN;  // bf16 x
  unsigned short* bufA  = wsh + hoff; hoff += (size_t)NNODE * HID;  // bf16 activations
  unsigned short* bufB  = wsh + hoff; hoff += (size_t)NNODE * HID;
  unsigned short* bufC  = wsh + hoff; hoff += (size_t)NNODE * HID;
  unsigned short* Wt0   = wsh + hoff; hoff += (size_t)FIN * HID;    // enc0^T [128 x 512]
  unsigned short* Wt1   = wsh + hoff; hoff += (size_t)HID * HID;
  unsigned short* Wt2   = wsh + hoff; hoff += (size_t)HID * HID;
  unsigned short* Wt3   = wsh + hoff; hoff += (size_t)HID * HID;
  unsigned short* Wt4   = wsh + hoff; hoff += (size_t)HID * HID;
  unsigned short* Wt5   = wsh + hoff; hoff += (size_t)HID * FIN;    // ad1^T [512 x 128]

  const int T = 256;

  // degrees (self-loop contributes 1) and D^-1/2
  k_fill      <<<(NNODE + T - 1) / T, T, 0, stream>>>(deg, 1.0f, NNODE);
  k_deg_edges <<<(E + T - 1) / T,     T, 0, stream>>>(dst, deg, E);
  k_dinv      <<<(NNODE + T - 1) / T, T, 0, stream>>>(deg, dinv, NNODE);

  // one-time conversions: x -> bf16, weights -> transposed bf16 [Fout x K]
  k_cvt_bf16 <<<(NNODE * FIN / 4 + T - 1) / T, T, 0, stream>>>(x, xb, NNODE * FIN / 4);
  k_transpose_bf16 <<<(FIN * HID + T - 1) / T, T, 0, stream>>>(W_enc0, Wt0, FIN, HID);
  k_transpose_bf16 <<<(HID * HID + T - 1) / T, T, 0, stream>>>(W_enc1, Wt1, HID, HID);
  k_transpose_bf16 <<<(HID * HID + T - 1) / T, T, 0, stream>>>(W_sd0,  Wt2, HID, HID);
  k_transpose_bf16 <<<(HID * HID + T - 1) / T, T, 0, stream>>>(W_sd1,  Wt3, HID, HID);
  k_transpose_bf16 <<<(HID * HID + T - 1) / T, T, 0, stream>>>(W_ad0,  Wt4, HID, HID);
  k_transpose_bf16 <<<(HID * FIN + T - 1) / T, T, 0, stream>>>(W_ad1,  Wt5, HID, FIN);

  // GCN conv: GEMM (bf16 WMMA) -> normalize/aggregate -> bias+ReLU (to bf16)
  auto conv = [&](const unsigned short* hin, int Fin, const unsigned short* Wt, int Fout,
                  const float* bias, unsigned short* outb) {
    dim3 g(Fout / 64, NNODE / 128);
    k_gemm_abT_bf16 <<<g, 256, 0, stream>>>(hin, Wt, tmp, NNODE, Fout, Fin, Fout);
    int total = NNODE * Fout;
    int logF  = (Fout == 512) ? 9 : 7;
    k_agg_self  <<<(total + T - 1) / T, T, 0, stream>>>(tmp, dinv, agg, total, logF);
    k_agg_edges <<<(E + 7) / 8, 256, 0, stream>>>(tmp, src, dst, dinv, agg, E, Fout);
    k_bias_relu_bf16 <<<(total + T - 1) / T, T, 0, stream>>>(agg, bias, outb, total, Fout - 1);
  };

  // encoder
  conv(xb,   FIN, Wt0, HID, b_enc0, bufA);       // z0
  conv(bufA, HID, Wt1, HID, b_enc1, bufB);       // z   (kept: feeds both decoders)
  // structure decoder
  conv(bufB, HID, Wt2, HID, b_sd0, bufC);        // sz0
  conv(bufC, HID, Wt3, HID, b_sd1, bufA);        // sz  (bf16, feeds sz@sz^T)
  // attribute decoder
  conv(bufB, HID, Wt4, HID, b_ad0, bufC);        // a

  // last conv writes f32 attr_recon straight into d_out tail
  float* attr = (float*)d_out + (size_t)NNODE * NNODE;
  {
    dim3 g(FIN / 64, NNODE / 128);
    k_gemm_abT_bf16 <<<g, 256, 0, stream>>>(bufC, Wt5, tmp, NNODE, FIN, HID, FIN);
    int total = NNODE * FIN;
    k_agg_self  <<<(total + T - 1) / T, T, 0, stream>>>(tmp, dinv, attr, total, 9);
    k_agg_edges <<<(E + 7) / 8, 256, 0, stream>>>(tmp, src, dst, dinv, attr, E, FIN);
    k_bias_relu_f32 <<<(total + T - 1) / T, T, 0, stream>>>(attr, b_ad1, total, FIN - 1);
  }

  // structure reconstruction: stru = sz @ sz^T (B^T columns == contiguous sz rows)
  dim3 gs(NNODE / 64, NNODE / 128);
  k_gemm_abT_bf16 <<<gs, 256, 0, stream>>>(bufA, bufA, (float*)d_out,
                                           NNODE, NNODE, HID, NNODE);
}
